// MONET_9053791060400
// MI455X (gfx1250) — matrix operand order repaired and checked by reading, not compiled, not run on hardware
//
#include <hip/hip_runtime.h>
#include <math.h>

// ---------------- types ----------------
typedef __attribute__((ext_vector_type(16))) __bf16 v16bf;
typedef __attribute__((ext_vector_type(8)))  float  v8f;

#define N_USERS 100000
#define N_ITEMS 50000
#define N_TOTAL (N_USERS + N_ITEMS)
#define NNZ_CNT 6400000
#define DIM     64
#define ALPHA_C 0.3f

// ---------------------------------------------------------------------------
// Convert W [K x 64] fp32 -> transposed bf16 hi/lo [64 x K]
// ---------------------------------------------------------------------------
__global__ __launch_bounds__(256) void monet_wconv(
    const float* __restrict__ W, __bf16* __restrict__ hi,
    __bf16* __restrict__ lo, int K)
{
    int idx = blockIdx.x * blockDim.x + threadIdx.x;
    if (idx >= K * DIM) return;
    int k = idx / DIM;
    int n = idx % DIM;
    float x  = W[idx];
    __bf16 h = (__bf16)x;
    hi[(long)n * K + k] = h;
    lo[(long)n * K + k] = (__bf16)(x - (float)h);
}

// ---------------------------------------------------------------------------
// Fused projection: out[M x 64] = l2norm_rows(A[M x K] @ W + b)
// One wave -> 16 rows x 64 cols. bf16 WMMA with 2-term operand split
// (hi*hi + hi*lo + lo*hi) for ~fp32 accuracy. M must be a multiple of 16.
// Loop body is shaped as: issue ALL loads (A + 8 B vectors) -> single wait
// -> 12 back-to-back WMMAs, so the XDL pipe runs uninterrupted bursts.
// ---------------------------------------------------------------------------
__global__ __launch_bounds__(256) void monet_proj_wmma(
    const float*  __restrict__ A, int K,
    const __bf16* __restrict__ BtHi, const __bf16* __restrict__ BtLo,
    const float*  __restrict__ bias, float* __restrict__ out, int M)
{
    const int wave = threadIdx.x >> 5;
    const int lane = threadIdx.x & 31;
    const int half = lane >> 4;        // which 16-lane half
    const int nl   = lane & 15;
    const int tile = blockIdx.x * (blockDim.x >> 5) + wave;
    if (tile * 16 >= M) return;        // wave-uniform exit (EXEC stays all-1s)

    const float* __restrict__ arow =
        A + ((long)tile * 16 + nl) * (long)K + half * 16;

    // Hoisted B row pointers (row n of W^T = column n of W), one per N-tile.
    const long brow = (long)nl * K + half * 16;
    const __bf16* __restrict__ pbh0 = BtHi + brow;
    const __bf16* __restrict__ pbh1 = BtHi + brow + (long)16 * K;
    const __bf16* __restrict__ pbh2 = BtHi + brow + (long)32 * K;
    const __bf16* __restrict__ pbh3 = BtHi + brow + (long)48 * K;
    const __bf16* __restrict__ pbl0 = BtLo + brow;
    const __bf16* __restrict__ pbl1 = BtLo + brow + (long)16 * K;
    const __bf16* __restrict__ pbl2 = BtLo + brow + (long)32 * K;
    const __bf16* __restrict__ pbl3 = BtLo + brow + (long)48 * K;

    v8f acc[4] = {};                   // 16x16 f32 accumulators, 4 N-tiles

    for (int k0 = 0; k0 < K; k0 += 32) {
        // ---- issue all loads for this K-step up front ----
        const float* p0 = arow + k0;
        float4 f0 = *(const float4*)(p0);
        float4 f1 = *(const float4*)(p0 + 4);
        float4 f2 = *(const float4*)(p0 + 8);
        float4 f3 = *(const float4*)(p0 + 12);
        __builtin_prefetch(p0 + 64, 0, 1);   // global_prefetch next chunks

        v16bf bh[4], bl[4];
        bh[0] = *(const v16bf*)(pbh0 + k0);
        bh[1] = *(const v16bf*)(pbh1 + k0);
        bh[2] = *(const v16bf*)(pbh2 + k0);
        bh[3] = *(const v16bf*)(pbh3 + k0);
        bl[0] = *(const v16bf*)(pbl0 + k0);
        bl[1] = *(const v16bf*)(pbl1 + k0);
        bl[2] = *(const v16bf*)(pbl2 + k0);
        bl[3] = *(const v16bf*)(pbl3 + k0);

        // ---- A fragment split-convert (k order used here, same as B:
        //      half 0 -> K 0-15, half 1 -> K 16-31) ----
        float buf[16] = { f0.x, f0.y, f0.z, f0.w,  f1.x, f1.y, f1.z, f1.w,
                          f2.x, f2.y, f2.z, f2.w,  f3.x, f3.y, f3.z, f3.w };
        v16bf a_hi, a_lo;
#pragma unroll
        for (int i = 0; i < 16; ++i) {
            __bf16 h = (__bf16)buf[i];
            a_hi[i] = h;
            a_lo[i] = (__bf16)(buf[i] - (float)h);
        }

        // ---- 12 back-to-back WMMAs ----
#pragma unroll
        for (int t = 0; t < 4; ++t) {
            acc[t] = __builtin_amdgcn_wmma_f32_16x16x32_bf16(
                         false, a_hi, false, bh[t], (short)0, acc[t], false, false);
            acc[t] = __builtin_amdgcn_wmma_f32_16x16x32_bf16(
                         false, a_lo, false, bh[t], (short)0, acc[t], false, false);
            acc[t] = __builtin_amdgcn_wmma_f32_16x16x32_bf16(
                         false, a_hi, false, bl[t], (short)0, acc[t], false, false);
        }
    }

    // ---- bias (per column) ----
    float bv[4];
#pragma unroll
    for (int t = 0; t < 4; ++t) bv[t] = bias[t * 16 + nl];
#pragma unroll
    for (int t = 0; t < 4; ++t)
#pragma unroll
        for (int j = 0; j < 8; ++j) acc[t][j] += bv[t];

    // ---- row L2 norm: C layout row(j) = j + 8*half, cols across 16 lanes ----
    float sc[8];
#pragma unroll
    for (int j = 0; j < 8; ++j) {
        float s = acc[0][j]*acc[0][j] + acc[1][j]*acc[1][j]
                + acc[2][j]*acc[2][j] + acc[3][j]*acc[3][j];
        s += __shfl_xor(s, 1);
        s += __shfl_xor(s, 2);
        s += __shfl_xor(s, 4);
        s += __shfl_xor(s, 8);
        sc[j] = 1.0f / fmaxf(sqrtf(s), 1e-12f);
    }

    // ---- store normalized rows ----
    float* __restrict__ orow = out + (long)tile * 16 * DIM + (long)half * 8 * DIM + nl;
#pragma unroll
    for (int j = 0; j < 8; ++j)
#pragma unroll
        for (int t = 0; t < 4; ++t)
            orow[(long)j * DIM + t * 16] = acc[t][j] * sc[j];
}

// ---------------------------------------------------------------------------
// out = ALPHA * in  (both channels, float4)
// ---------------------------------------------------------------------------
__global__ __launch_bounds__(256) void monet_scale2(
    const float4* __restrict__ inI, const float4* __restrict__ inT,
    float4* __restrict__ outI, float4* __restrict__ outT, long n4)
{
    long i = (long)blockIdx.x * blockDim.x + threadIdx.x;
    if (i >= n4) return;
    float4 a = inI[i];
    float4 b = inT[i];
    outI[i] = make_float4(ALPHA_C*a.x, ALPHA_C*a.y, ALPHA_C*a.z, ALPHA_C*a.w);
    outT[i] = make_float4(ALPHA_C*b.x, ALPHA_C*b.y, ALPHA_C*b.z, ALPHA_C*b.w);
}

// ---------------------------------------------------------------------------
// COO SpMM, both channels fused: one wave per nonzero.
// lanes 0-15 -> image channel, lanes 16-31 -> text channel; float4 per lane.
// Working set (4 x 38.4 MB) is L2-resident on MI455X (192 MB L2), so the
// random gathers and f32 atomics resolve in L2; only the 77 MB of COO
// indices per layer stream from HBM.
// ---------------------------------------------------------------------------
__global__ __launch_bounds__(256) void monet_spmm(
    const float* __restrict__ vals, const int* __restrict__ rows,
    const int* __restrict__ cols,
    const float* __restrict__ inI, const float* __restrict__ inT,
    float* __restrict__ outI, float* __restrict__ outT, int nnz)
{
    int e = blockIdx.x * (blockDim.x >> 5) + (threadIdx.x >> 5);
    if (e >= nnz) return;
    int lane = threadIdx.x & 31;
    int r = rows[e];           // wave-uniform -> scalar loads
    int c = cols[e];
    float v = vals[e];

    const float* __restrict__ in  = (lane < 16) ? inI  : inT;
    float*       __restrict__ out = (lane < 16) ? outI : outT;
    int q = (lane & 15) * 4;

    float4 x = *(const float4*)(in + (long)c * DIM + q);
    float* o = out + (long)r * DIM + q;
    atomicAdd(o + 0, v * x.x);
    atomicAdd(o + 1, v * x.y);
    atomicAdd(o + 2, v * x.z);
    atomicAdd(o + 3, v * x.w);
}

// ---------------------------------------------------------------------------
// Row-wise concat of the two channels into d_out (users then items — same
// row ordering as the ego buffers, so it is a pure layout transform).
// ---------------------------------------------------------------------------
__global__ __launch_bounds__(256) void monet_gather_out(
    const float* __restrict__ egoI, const float* __restrict__ egoT,
    float* __restrict__ out, long nrows)
{
    long i = (long)blockIdx.x * blockDim.x + threadIdx.x;
    if (i >= nrows * 32) return;            // 32 float4 per output row (128 f)
    long row = i >> 5;
    int  col = (int)(i & 31) << 2;
    const float* src = (col < 64) ? (egoI + row * DIM + col)
                                  : (egoT + row * DIM + (col - 64));
    float4 v = *(const float4*)src;
    *(float4*)(out + row * 128 + col) = v;
}

// ---------------------------------------------------------------------------
extern "C" void kernel_launch(void* const* d_in, const int* in_sizes, int n_in,
                              void* d_out, int out_size, void* d_ws, size_t ws_size,
                              hipStream_t stream)
{
    (void)in_sizes; (void)n_in; (void)out_size; (void)ws_size;

    const float* image_feats = (const float*)d_in[0];
    const float* text_feats  = (const float*)d_in[1];
    const float* image_pref  = (const float*)d_in[2];
    const float* text_pref   = (const float*)d_in[3];
    const float* W_img       = (const float*)d_in[4];
    const float* b_img       = (const float*)d_in[5];
    const float* W_txt       = (const float*)d_in[6];
    const float* b_txt       = (const float*)d_in[7];
    const float* adj_vals    = (const float*)d_in[8];
    const int*   adj_rows    = (const int*)d_in[9];
    const int*   adj_cols    = (const int*)d_in[10];
    float* out = (float*)d_out;

    const long NV = (long)N_TOTAL * DIM;         // floats per ego buffer
    float* egoA_i = (float*)d_ws;
    float* egoA_t = egoA_i + NV;
    float* egoB_i = egoA_t + NV;
    float* egoB_t = egoB_i + NV;
    __bf16* wi_hi = (__bf16*)(egoB_t + NV);
    __bf16* wi_lo = wi_hi + (long)4096 * DIM;
    __bf16* wt_hi = wi_lo + (long)4096 * DIM;
    __bf16* wt_lo = wt_hi + (long)384  * DIM;

    // 1) user preference rows -> top of ego buffers
    hipMemcpyAsync(egoA_i, image_pref, (size_t)N_USERS * DIM * sizeof(float),
                   hipMemcpyDeviceToDevice, stream);
    hipMemcpyAsync(egoA_t, text_pref,  (size_t)N_USERS * DIM * sizeof(float),
                   hipMemcpyDeviceToDevice, stream);

    // 2) weights -> transposed bf16 hi/lo
    monet_wconv<<<(4096 * DIM + 255) / 256, 256, 0, stream>>>(W_img, wi_hi, wi_lo, 4096);
    monet_wconv<<<(384  * DIM + 255) / 256, 256, 0, stream>>>(W_txt, wt_hi, wt_lo, 384);

    // 3) fused projection + bias + l2norm -> item rows of ego buffers
    {
        int tiles  = N_ITEMS / 16;               // 3125
        int blocks = (tiles + 7) / 8;
        monet_proj_wmma<<<blocks, 256, 0, stream>>>(
            image_feats, 4096, wi_hi, wi_lo, b_img,
            egoA_i + (long)N_USERS * DIM, N_ITEMS);
        monet_proj_wmma<<<blocks, 256, 0, stream>>>(
            text_feats, 384, wt_hi, wt_lo, b_txt,
            egoA_t + (long)N_USERS * DIM, N_ITEMS);
    }

    const long n4 = NV / 4;
    const int scale_blocks = (int)((n4 + 255) / 256);
    const int spmm_blocks  = (NNZ_CNT + 7) / 8;

    // 4) layer 1: B = ALPHA*A; B += spmm(A)
    monet_scale2<<<scale_blocks, 256, 0, stream>>>(
        (const float4*)egoA_i, (const float4*)egoA_t,
        (float4*)egoB_i, (float4*)egoB_t, n4);
    monet_spmm<<<spmm_blocks, 256, 0, stream>>>(
        adj_vals, adj_rows, adj_cols, egoA_i, egoA_t, egoB_i, egoB_t, NNZ_CNT);

    // 5) layer 2: A = ALPHA*B; A += spmm(B)
    monet_scale2<<<scale_blocks, 256, 0, stream>>>(
        (const float4*)egoB_i, (const float4*)egoB_t,
        (float4*)egoA_i, (float4*)egoA_t, n4);
    monet_spmm<<<spmm_blocks, 256, 0, stream>>>(
        adj_vals, adj_rows, adj_cols, egoB_i, egoB_t, egoA_i, egoA_t, NNZ_CNT);

    // 6) concat channels -> d_out
    {
        long total = (long)N_TOTAL * 32;
        monet_gather_out<<<(int)((total + 255) / 256), 256, 0, stream>>>(
            egoA_i, egoA_t, out, N_TOTAL);
    }
}
